// TransformerLayer_31628139167857
// MI455X (gfx1250) — matrix operand order repaired and checked by reading, not compiled, not run on hardware
//
#include <hip/hip_runtime.h>

// ---------------------------------------------------------------------------
// Types for CDNA5 WMMA (wave32, 16x16x32 bf16, f32 accumulate)
// ---------------------------------------------------------------------------
typedef __attribute__((ext_vector_type(16))) __bf16 v16bf;
typedef __attribute__((ext_vector_type(8)))  float  v8f;
typedef int v4i __attribute__((vector_size(4 * sizeof(int))));

struct Pack32 { uint4 lo, hi; };   // 32 bytes == one v16bf fragment

#if __has_builtin(__builtin_amdgcn_global_load_async_to_lds_b128)
#define HAS_ASYNC_LDS 1
#else
#define HAS_ASYNC_LDS 0
#endif

#if HAS_ASYNC_LDS
// Async DMA: global -> LDS, 16B per lane, tracked by ASYNCcnt (ISA ch10 / §4).
__device__ __forceinline__ void async_cp_b128(const unsigned short* g, unsigned short* l) {
  __builtin_amdgcn_global_load_async_to_lds_b128(
      (__attribute__((address_space(1))) v4i*)g,
      (__attribute__((address_space(3))) v4i*)l, 0, 0);
}
__device__ __forceinline__ void wait_async0() {
#if __has_builtin(__builtin_amdgcn_s_wait_asynccnt)
  __builtin_amdgcn_s_wait_asynccnt(0);
#else
  asm volatile("s_wait_asynccnt 0x0" ::: "memory");
#endif
}
#endif

// RNE float -> bf16 (as raw ushort)
__device__ __forceinline__ unsigned short f2bf(float f) {
  unsigned u = __builtin_bit_cast(unsigned, f);
  return (unsigned short)((u + 0x7fffu + ((u >> 16) & 1u)) >> 16);
}

// A-matrix 16x32 bf16 fragment:
//   M=lane&15, elems 0..15 = K(2*kbase+0..15)
//   kbase = (lane>=16) ? 8 : 0. `row` points at the lane's row in LDS.
__device__ __forceinline__ v16bf ld_frag_a(const unsigned short* row, int kbase) {
  const unsigned short* p = row + 2 * kbase;
  Pack32 t;
  t.lo = *(const uint4*)(p);        // K = 2*kbase .. 2*kbase+7
  t.hi = *(const uint4*)(p + 8);    // K = 2*kbase+8 .. 2*kbase+15
  return __builtin_bit_cast(v16bf, t);
}

// B-matrix 32x16 bf16 fragment: lane -> column N=lane&15; element i -> K=kgrp+i,
// kgrp=(lane>>4)*16. With B stored [n][k] this is a contiguous 32-byte load.
__device__ __forceinline__ v16bf ld_frag_b(const unsigned short* p) {
  Pack32 t;
  t.lo = *(const uint4*)(p);
  t.hi = *(const uint4*)(p + 8);
  return __builtin_bit_cast(v16bf, t);
}

// ---------------------------------------------------------------------------
// One-time weight prep: fp32 [K][N] -> bf16 [N][K] (tiled LDS transpose).
// ---------------------------------------------------------------------------
__global__ __launch_bounds__(256)
void k_cvt_bf16_t(const float* __restrict__ src, unsigned short* __restrict__ dst,
                  int K, int N) {
  __shared__ float tile[32][33];
  const int k0 = blockIdx.y * 32, n0 = blockIdx.x * 32;
  const int tx = threadIdx.x & 31, ty = threadIdx.x >> 5;   // 32 x 8
#pragma unroll
  for (int i = ty; i < 32; i += 8)
    tile[i][tx] = src[(long)(k0 + i) * N + n0 + tx];
  __syncthreads();
#pragma unroll
  for (int i = ty; i < 32; i += 8)
    dst[(long)(n0 + i) * K + k0 + tx] = f2bf(tile[tx][i]);
}

// ---------------------------------------------------------------------------
// Row LayerNorm (H=2048), fp32 in -> bf16 out. One 256-thread block per row.
// ---------------------------------------------------------------------------
__global__ __launch_bounds__(256)
void k_layernorm_bf16(const float* __restrict__ x, const float* __restrict__ w,
                      const float* __restrict__ b, unsigned short* __restrict__ out,
                      int H) {
  const int row = blockIdx.x;
  const float* xr = x + (long)row * H;
  float s = 0.f, s2 = 0.f;
  for (int i = threadIdx.x; i < H; i += 256) { float v = xr[i]; s += v; s2 += v * v; }
#pragma unroll
  for (int m = 16; m >= 1; m >>= 1) { s += __shfl_xor(s, m, 32); s2 += __shfl_xor(s2, m, 32); }
  __shared__ float rs[8], rs2[8];
  const int wv = threadIdx.x >> 5;
  if ((threadIdx.x & 31) == 0) { rs[wv] = s; rs2[wv] = s2; }
  __syncthreads();
  float tot = 0.f, tot2 = 0.f;
#pragma unroll
  for (int i = 0; i < 8; ++i) { tot += rs[i]; tot2 += rs2[i]; }
  const float mu = tot / H;
  const float rstd = rsqrtf(tot2 / H - mu * mu + 1e-5f);
  unsigned short* orow = out + (long)row * H;
  for (int i = threadIdx.x; i < H; i += 256)
    orow[i] = f2bf((xr[i] - mu) * rstd * w[i] + b[i]);
}

// ---------------------------------------------------------------------------
// WMMA bf16 GEMM: C(MxN) = A(MxK) @ Bt(NxK)^T + bias, epilogue by mode:
//   mode 0: outb[m*N+n] = bf16((acc+bias)*scale)     (Q/K projections)
//   mode 1: outb[m*N+n] = bf16(gelu(acc+bias))       (MLP up-proj, exact erf)
//   mode 2: outf[m*N+n] = residual + acc + bias      (O-proj / MLP down-proj)
//   mode 3: outb[n*M+m] = bf16((acc+bias)*scale)     (V projection, transposed)
// Block 256 thr = 8 waves (2x4); tile 128x256x32; wave computes 64x64 (4x4
// frags = 16 WMMA / K-step). Double-buffered LDS; tile k+1 is fetched by the
// async global->LDS DMA (ASYNCcnt) while WMMA runs on tile k.
// ---------------------------------------------------------------------------
#define GBM 128
#define GBN 256
#define GBK 32
#define GLDS 40            // padded row stride (ushorts); 80B rows keep 16B align
#define ASZ (GBM * GLDS)
#define BSZ (GBN * GLDS)

__global__ __launch_bounds__(256)
void k_gemm_bf16(const unsigned short* __restrict__ A,
                 const unsigned short* __restrict__ Bt,
                 const float* __restrict__ bias,
                 const float* __restrict__ residual,
                 unsigned short* __restrict__ outb,
                 float* __restrict__ outf,
                 int M, int N, int K, int mode, float scale) {
  __shared__ unsigned short As[2 * ASZ];   // [buf][m][k]
  __shared__ unsigned short Bs[2 * BSZ];   // [buf][n][k]
  const int tid  = threadIdx.x;
  const int lane = tid & 31;
  const int half = lane >> 4;
  const int l16  = lane & 15;
  const int wid  = tid >> 5;
  const int wm   = (wid >> 2) * 64;    // 2 waves along M
  const int wn   = (wid & 3) * 64;     // 4 waves along N
  const long m0  = (long)blockIdx.y * GBM;
  const long n0  = (long)blockIdx.x * GBN;
  const int kbase = half * 8;

  v8f acc[4][4] = {};

#if HAS_ASYNC_LDS
  // A tile 128x32 = 2 b128/thr, B tile 256x32 = 4 b128/thr
  auto issue_tiles = [&](int kb, int buf) {
    unsigned short* as = As + buf * ASZ;
    unsigned short* bs = Bs + buf * BSZ;
#pragma unroll
    for (int it = 0; it < 2; ++it) {
      int c = tid + it * 256;
      async_cp_b128(A + (m0 + (c >> 2)) * K + kb + (c & 3) * 8,
                    as + (c >> 2) * GLDS + (c & 3) * 8);
    }
#pragma unroll
    for (int it = 0; it < 4; ++it) {
      int c = tid + it * 256;
      async_cp_b128(Bt + (n0 + (c >> 2)) * K + kb + (c & 3) * 8,
                    bs + (c >> 2) * GLDS + (c & 3) * 8);
    }
  };
  issue_tiles(0, 0);
  wait_async0();
  __syncthreads();
#else
  uint4 ra[2], rb[4];
  auto load_tiles = [&](int kb) {
#pragma unroll
    for (int it = 0; it < 2; ++it) {
      int c = tid + it * 256;
      ra[it] = *(const uint4*)(A + (m0 + (c >> 2)) * K + kb + (c & 3) * 8);
    }
#pragma unroll
    for (int it = 0; it < 4; ++it) {
      int c = tid + it * 256;
      rb[it] = *(const uint4*)(Bt + (n0 + (c >> 2)) * K + kb + (c & 3) * 8);
    }
  };
  auto store_tiles = [&](int buf) {
    unsigned short* as = As + buf * ASZ;
    unsigned short* bs = Bs + buf * BSZ;
#pragma unroll
    for (int it = 0; it < 2; ++it) {
      int c = tid + it * 256;
      *(uint4*)(as + (c >> 2) * GLDS + (c & 3) * 8) = ra[it];
    }
#pragma unroll
    for (int it = 0; it < 4; ++it) {
      int c = tid + it * 256;
      *(uint4*)(bs + (c >> 2) * GLDS + (c & 3) * 8) = rb[it];
    }
  };
  load_tiles(0);
  store_tiles(0);
  __syncthreads();
#endif

  int buf = 0;
  for (int kb = 0; kb < K; kb += GBK) {
    const bool more = (kb + GBK) < K;
#if HAS_ASYNC_LDS
    if (more) issue_tiles(kb + GBK, buf ^ 1);   // DMA next tile under the WMMAs
#else
    if (more) load_tiles(kb + GBK);
#endif

    const unsigned short* as = As + buf * ASZ;
    const unsigned short* bs = Bs + buf * BSZ;
    v16bf af[4];
#pragma unroll
    for (int fm = 0; fm < 4; ++fm)
      af[fm] = ld_frag_a(as + (wm + fm * 16 + l16) * GLDS, kbase);
#pragma unroll
    for (int fn = 0; fn < 4; ++fn) {
      v16bf bf = ld_frag_b(bs + (wn + fn * 16 + l16) * GLDS + half * 16);
#pragma unroll
      for (int fm = 0; fm < 4; ++fm)
        acc[fm][fn] = __builtin_amdgcn_wmma_f32_16x16x32_bf16(
            false, af[fm], false, bf, (short)0, acc[fm][fn], false, false);
    }

#if HAS_ASYNC_LDS
    if (more) wait_async0();
#else
    if (more) store_tiles(buf ^ 1);
#endif
    __syncthreads();
    buf ^= 1;
  }

  // ---- epilogue ----
  // C/D layout: VGPR r -> row (r + 8*half), col = l16. Mode branch hoisted out
  // of the loops; one row/col base pointer per (fm,r)/(fn) to keep VGPR
  // pressure flat (avoids spilling the 128 accumulator VGPRs).
  float bias4[4];
#pragma unroll
  for (int fn = 0; fn < 4; ++fn) bias4[fn] = bias[n0 + wn + fn * 16 + l16];

  if (mode == 0) {
#pragma unroll
    for (int fm = 0; fm < 4; ++fm)
#pragma unroll
      for (int r = 0; r < 8; ++r) {
        unsigned short* prow =
            outb + (m0 + wm + fm * 16 + half * 8 + r) * (long)N + n0 + wn + l16;
#pragma unroll
        for (int fn = 0; fn < 4; ++fn)
          prow[fn * 16] = f2bf((acc[fm][fn][r] + bias4[fn]) * scale);
      }
  } else if (mode == 1) {
#pragma unroll
    for (int fm = 0; fm < 4; ++fm)
#pragma unroll
      for (int r = 0; r < 8; ++r) {
        unsigned short* prow =
            outb + (m0 + wm + fm * 16 + half * 8 + r) * (long)N + n0 + wn + l16;
#pragma unroll
        for (int fn = 0; fn < 4; ++fn) {
          float v = acc[fm][fn][r] + bias4[fn];
          prow[fn * 16] = f2bf(0.5f * v * (1.f + erff(v * 0.70710678118f)));
        }
      }
  } else if (mode == 2) {
#pragma unroll
    for (int fm = 0; fm < 4; ++fm)
#pragma unroll
      for (int r = 0; r < 8; ++r) {
        const long offr = (m0 + wm + fm * 16 + half * 8 + r) * (long)N + n0 + wn + l16;
        const float* rrow = residual + offr;
        float* orow = outf + offr;
#pragma unroll
        for (int fn = 0; fn < 4; ++fn)
          orow[fn * 16] = rrow[fn * 16] + acc[fm][fn][r] + bias4[fn];
      }
  } else {  // mode 3: transposed bf16 store (V projection)
#pragma unroll
    for (int fn = 0; fn < 4; ++fn) {
      unsigned short* pcol =
          outb + (n0 + wn + fn * 16 + l16) * (long)M + m0 + wm + half * 8;
#pragma unroll
      for (int fm = 0; fm < 4; ++fm)
#pragma unroll
        for (int r = 0; r < 8; ++r)
          pcol[fm * 16 + r] = f2bf((acc[fm][fn][r] + bias4[fn]) * scale);
    }
  }
}

// ---------------------------------------------------------------------------
// Flash attention (causal), bf16 in/out, fp32 online softmax.
// Grid: (S/64, NH). Block: 128 thr = 4 waves; wave owns 16 query rows.
// Q pre-scaled by 1/sqrt(HD). V arrives pre-transposed (Vt: [H][S]). K/V tiles
// ping-pong in LDS via async DMA (or reg-staging fallback). P aliases Q's tile.
// ---------------------------------------------------------------------------
#define FLD 72   // padded row stride (144B rows, 16B aligned)
#define FTS (64 * FLD)
#if HAS_ASYNC_LDS
#define FNB 2
#else
#define FNB 1
#endif

__global__ __launch_bounds__(128)
void k_flash_attn(const unsigned short* __restrict__ Q,
                  const unsigned short* __restrict__ Kk,
                  const unsigned short* __restrict__ Vt,
                  unsigned short* __restrict__ O,
                  int S, int H) {
  __shared__ unsigned short QPs[FTS];        // [q][d] then reused as P [q][k]
  __shared__ unsigned short Ks [FNB * FTS];  // [k][d]
  __shared__ unsigned short Vst[FNB * FTS];  // [d][k]
  const int tid  = threadIdx.x;
  const int lane = tid & 31;
  const int half = lane >> 4;
  const int l16  = lane & 15;
  const int wid  = tid >> 5;            // 0..3
  const int qt   = blockIdx.x;
  const long q0  = (long)qt * 64;
  const long hoff = (long)blockIdx.y * 64;
  const int kbase = half * 8;

#if HAS_ASYNC_LDS
  auto issue_kv = [&](long k0, int buf) {
#pragma unroll
    for (int it = 0; it < 2; ++it) {
      int c = tid + it * 128;
      async_cp_b128(Kk + (k0 + (c >> 2)) * H + hoff + (c & 3) * 8,
                    Ks + buf * FTS + (c >> 2) * FLD + (c & 3) * 8);
      async_cp_b128(Vt + (hoff + (c >> 2)) * S + k0 + (c & 3) * 8,
                    Vst + buf * FTS + (c >> 2) * FLD + (c & 3) * 8);
    }
  };
#else
  uint4 rk[2], rv[2];
  auto load_kv = [&](long k0) {
#pragma unroll
    for (int it = 0; it < 2; ++it) {
      int c = tid + it * 128;
      rk[it] = *(const uint4*)(Kk + (k0 + (c >> 2)) * H + hoff + (c & 3) * 8);
      rv[it] = *(const uint4*)(Vt + (hoff + (c >> 2)) * S + k0 + (c & 3) * 8);
    }
  };
  auto store_kv = [&]() {
#pragma unroll
    for (int it = 0; it < 2; ++it) {
      int c = tid + it * 128;
      *(uint4*)(Ks  + (c >> 2) * FLD + (c & 3) * 8) = rk[it];
      *(uint4*)(Vst + (c >> 2) * FLD + (c & 3) * 8) = rv[it];
    }
  };
#endif

  // stage Q tile; start fetching K/V tile 0
#pragma unroll
  for (int it = 0; it < 2; ++it) {
    int c = tid + it * 128;
    *(uint4*)(QPs + (c >> 2) * FLD + (c & 3) * 8) =
        *(const uint4*)(Q + (q0 + (c >> 2)) * H + hoff + (c & 3) * 8);
  }
#if HAS_ASYNC_LDS
  issue_kv(0, 0);
  __syncthreads();            // Q tile visible
#else
  load_kv(0);
  __syncthreads();
#endif
  const v16bf aq0 = ld_frag_a(QPs + (wid * 16 + l16) * FLD, kbase);        // d 0..31
  const v16bf aq1 = ld_frag_a(QPs + (wid * 16 + l16) * FLD + 32, kbase);   // d 32..63
#if HAS_ASYNC_LDS
  wait_async0();
  __syncthreads();            // Q frags in regs (QPs reusable as P); KV0 visible
#else
  __syncthreads();
  store_kv();
  __syncthreads();
#endif

  float mi[8], li[8];
  v8f o[4] = {};
#pragma unroll
  for (int r = 0; r < 8; ++r) { mi[r] = -3.0e38f; li[r] = 0.f; }

  int buf = 0;
  for (int kt = 0; kt <= qt; ++kt) {
    const bool more = kt < qt;
#if HAS_ASYNC_LDS
    if (more) issue_kv((long)(kt + 1) * 64, buf ^ 1);   // DMA next K/V tile
#else
    if (more) load_kv((long)(kt + 1) * 64);
#endif
    const unsigned short* ks = Ks + buf * FTS;
    const unsigned short* vs = Vst + buf * FTS;

    // S(16x64) = Q_w(16x64) @ K^T
    v8f s[4] = {};
#pragma unroll
    for (int fn = 0; fn < 4; ++fn) {
      const unsigned short* kr = ks + (fn * 16 + l16) * FLD + half * 16;
      s[fn] = __builtin_amdgcn_wmma_f32_16x16x32_bf16(false, aq0, false, ld_frag_b(kr),
                                                      (short)0, s[fn], false, false);
      s[fn] = __builtin_amdgcn_wmma_f32_16x16x32_bf16(false, aq1, false, ld_frag_b(kr + 32),
                                                      (short)0, s[fn], false, false);
    }
    if (kt == qt) {  // causal (additive -1e9, matching the reference mask)
#pragma unroll
      for (int fn = 0; fn < 4; ++fn)
#pragma unroll
        for (int r = 0; r < 8; ++r)
          if (fn * 16 + l16 > wid * 16 + r + half * 8) s[fn][r] += -1.0e9f;
    }

    // online softmax; row m lives in VGPR r across one 16-lane half
    float rowm[8], al[8], rsum[8];
#pragma unroll
    for (int r = 0; r < 8; ++r)
      rowm[r] = fmaxf(fmaxf(s[0][r], s[1][r]), fmaxf(s[2][r], s[3][r]));
#pragma unroll
    for (int mk = 8; mk >= 1; mk >>= 1)
#pragma unroll
      for (int r = 0; r < 8; ++r)
        rowm[r] = fmaxf(rowm[r], __shfl_xor(rowm[r], mk, 32));
#pragma unroll
    for (int r = 0; r < 8; ++r) {
      float mn = fmaxf(mi[r], rowm[r]);
      al[r] = __expf(mi[r] - mn);
      mi[r] = mn;
      rsum[r] = 0.f;
    }
#pragma unroll
    for (int fn = 0; fn < 4; ++fn)
#pragma unroll
      for (int r = 0; r < 8; ++r) {
        float p = __expf(s[fn][r] - mi[r]);
        s[fn][r] = p;
        rsum[r] += p;
      }
#pragma unroll
    for (int mk = 8; mk >= 1; mk >>= 1)
#pragma unroll
      for (int r = 0; r < 8; ++r)
        rsum[r] += __shfl_xor(rsum[r], mk, 32);
#pragma unroll
    for (int r = 0; r < 8; ++r) li[r] = li[r] * al[r] + rsum[r];
#pragma unroll
    for (int fn = 0; fn < 4; ++fn)
#pragma unroll
      for (int r = 0; r < 8; ++r) o[fn][r] *= al[r];

    // P -> LDS (bf16): reshape C-layout into A-fragment layout (per-wave rows;
    // same-wave DS ordering guarantees store->load visibility)
#pragma unroll
    for (int fn = 0; fn < 4; ++fn)
#pragma unroll
      for (int r = 0; r < 8; ++r)
        QPs[(wid * 16 + r + half * 8) * FLD + fn * 16 + l16] = f2bf(s[fn][r]);
    v16bf ap0 = ld_frag_a(QPs + (wid * 16 + l16) * FLD, kbase);
    v16bf ap1 = ld_frag_a(QPs + (wid * 16 + l16) * FLD + 32, kbase);
#pragma unroll
    for (int fn = 0; fn < 4; ++fn) {
      const unsigned short* vr = vs + (fn * 16 + l16) * FLD + half * 16;
      o[fn] = __builtin_amdgcn_wmma_f32_16x16x32_bf16(false, ap0, false, ld_frag_b(vr),
                                                      (short)0, o[fn], false, false);
      o[fn] = __builtin_amdgcn_wmma_f32_16x16x32_bf16(false, ap1, false, ld_frag_b(vr + 32),
                                                      (short)0, o[fn], false, false);
    }

#if HAS_ASYNC_LDS
    if (more) wait_async0();
    __syncthreads();
    buf ^= 1;
#else
    __syncthreads();             // all waves done reading Ks/Vst
    if (more) store_kv();
    __syncthreads();
#endif
  }

  // output: one row pointer + one reciprocal per row
#pragma unroll
  for (int r = 0; r < 8; ++r) {
    unsigned short* prow = O + (q0 + wid * 16 + half * 8 + r) * (long)H + hoff + l16;
    const float inv = 1.f / li[r];
#pragma unroll
    for (int fn = 0; fn < 4; ++fn)
      prow[fn * 16] = f2bf(o[fn][r] * inv);
  }
}

// ---------------------------------------------------------------------------
// Host-side orchestration
// ---------------------------------------------------------------------------
extern "C" void kernel_launch(void* const* d_in, const int* in_sizes, int n_in,
                              void* d_out, int out_size, void* d_ws, size_t ws_size,
                              hipStream_t stream) {
  (void)in_sizes; (void)n_in; (void)out_size; (void)ws_size;
  constexpr int S = 2048, H = 2048, NH = 32, DFF = 8192;

  const float* hs   = (const float*)d_in[0];
  const float* ln1w = (const float*)d_in[2];
  const float* ln1b = (const float*)d_in[3];
  const float* wq = (const float*)d_in[4];  const float* bq = (const float*)d_in[5];
  const float* wk = (const float*)d_in[6];  const float* bk = (const float*)d_in[7];
  const float* wv = (const float*)d_in[8];  const float* bv = (const float*)d_in[9];
  const float* wo = (const float*)d_in[10]; const float* bo = (const float*)d_in[11];
  const float* w1 = (const float*)d_in[12]; const float* b1 = (const float*)d_in[13];
  const float* w2 = (const float*)d_in[14]; const float* b2 = (const float*)d_in[15];
  const float* ln2w = (const float*)d_in[16];
  const float* ln2b = (const float*)d_in[17];
  float* out = (float*)d_out;

  size_t off = 0;
  auto alloc = [&](size_t bytes) -> void* {
    void* p = (char*)d_ws + off;
    off += (bytes + 255) & ~(size_t)255;
    return p;
  };
  unsigned short* wqb  = (unsigned short*)alloc((size_t)H * H * 2);     // [N][K] bf16
  unsigned short* wkb  = (unsigned short*)alloc((size_t)H * H * 2);
  unsigned short* wvb  = (unsigned short*)alloc((size_t)H * H * 2);
  unsigned short* wob  = (unsigned short*)alloc((size_t)H * H * 2);
  unsigned short* w1b  = (unsigned short*)alloc((size_t)H * DFF * 2);
  unsigned short* w2b  = (unsigned short*)alloc((size_t)DFF * H * 2);
  unsigned short* ln1o = (unsigned short*)alloc((size_t)S * H * 2);
  unsigned short* Qb   = (unsigned short*)alloc((size_t)S * H * 2);
  unsigned short* Kb   = (unsigned short*)alloc((size_t)S * H * 2);
  unsigned short* Vtb  = (unsigned short*)alloc((size_t)S * H * 2);     // [H][S] bf16
  unsigned short* attn = (unsigned short*)alloc((size_t)S * H * 2);
  float*          x1   = (float*)alloc((size_t)S * H * 4);
  unsigned short* ln2o = (unsigned short*)alloc((size_t)S * H * 2);
  unsigned short* h1b  = (unsigned short*)alloc((size_t)S * DFF * 2);

  auto cvtT = [&](const float* s, unsigned short* d, int K_, int N_) {
    k_cvt_bf16_t<<<dim3(N_ / 32, K_ / 32), dim3(256), 0, stream>>>(s, d, K_, N_);
  };
  cvtT(wq, wqb, H, H);
  cvtT(wk, wkb, H, H);
  cvtT(wv, wvb, H, H);
  cvtT(wo, wob, H, H);
  cvtT(w1, w1b, H, DFF);
  cvtT(w2, w2b, DFF, H);

  auto gemm = [&](const unsigned short* A, const unsigned short* Bt, const float* bias,
                  const float* res, unsigned short* ob, float* of,
                  int M, int N, int K, int mode, float scale) {
    k_gemm_bf16<<<dim3(N / GBN, M / GBM), dim3(256), 0, stream>>>(
        A, Bt, bias, res, ob, of, M, N, K, mode, scale);
  };

  // --- attention block ---
  k_layernorm_bf16<<<dim3(S), dim3(256), 0, stream>>>(hs, ln1w, ln1b, ln1o, H);
  gemm(ln1o, wqb, bq, nullptr, Qb,  nullptr, S, H, H, 0, 0.125f);  // 1/sqrt(64) folded
  gemm(ln1o, wkb, bk, nullptr, Kb,  nullptr, S, H, H, 0, 1.0f);
  gemm(ln1o, wvb, bv, nullptr, Vtb, nullptr, S, H, H, 3, 1.0f);    // V stored [H][S]
  k_flash_attn<<<dim3(S / 64, NH), dim3(128), 0, stream>>>(Qb, Kb, Vtb, attn, S, H);
  gemm(attn, wob, bo, hs, nullptr, x1, S, H, H, 2, 1.0f);          // x1 = hs + attn@wo+bo

  // --- MLP block ---
  k_layernorm_bf16<<<dim3(S), dim3(256), 0, stream>>>(x1, ln2w, ln2b, ln2o, H);
  gemm(ln2o, w1b, b1, nullptr, h1b, nullptr, S, DFF, H, 1, 1.0f);  // gelu
  gemm(h1b, w2b, b2, x1, nullptr, out, S, H, DFF, 2, 1.0f);        // out = x1 + h1@w2+b2
}
